// BiLSTM_MEMM_25512105738565
// MI455X (gfx1250) — compile-verified, matrix operand order checked by reading, not run on hardware
//
#include <hip/hip_runtime.h>
#include <hip/hip_bf16.h>
#include <stdint.h>

typedef __attribute__((ext_vector_type(16))) __bf16 v16bf;
typedef __attribute__((ext_vector_type(8)))  float  v8f;
typedef __attribute__((ext_vector_type(4)))  unsigned int u32x4;
typedef __attribute__((ext_vector_type(8)))  int  i32x8;
typedef __attribute__((ext_vector_type(4)))  int  i32x4;

#define L_SEQ 4096
#define E_DIM 1024
#define HDIM  512
#define G4    2048   // 4*HD
#define NTAGS 6
#define TAG_START 4
#define TAG_STOP  5

// persistent LSTM partitioning: 16 WGs per direction (one WGP each, cluster-sized)
#define NWG          16
#define CELLS_PER_WG 32              // 512 / 16
#define ROWS_PER_WG  128             // 4 gates * 32 cells
#define LSTM_SMEM    (ROWS_PER_WG*HDIM*2 + HDIM*4 + ROWS_PER_WG*4 + CELLS_PER_WG*4)

// Toolchain hedge: the therock/amdgpu-toolchain headers (6-arg TDM builtin) ship
// amd_gfx1250_TDM.h; ROCm 7.2 (5-arg TDM builtin) does not.
#if __has_include(<hip/amd_detail/amd_gfx1250_TDM.h>)
#define TDM_6ARG 1
#else
#define TDM_6ARG 0
#endif

// ---------- helpers ----------
static __device__ __forceinline__ unsigned short f32_bf16(float f) {
  unsigned int u = __float_as_uint(f);
  u += 0x7FFFu + ((u >> 16) & 1u);           // round-to-nearest-even
  return (unsigned short)(u >> 16);
}
static __device__ __forceinline__ float sigmoidf(float x) { return 1.0f / (1.0f + __expf(-x)); }

// ---------- Tensor Data Mover: contiguous 1-D global -> LDS copy ----------
// D# packed per CDNA5 ISA ch.8: group0 = {count/flags, lds_addr, global_addr, type=2},
// group1 = {mask=0, data_size=4B, tensor_dim0=tile_dim0=n, dim1=1}. Groups 2/3 zero
// (<=2D tensor). Issue from ONE wave only (TDM ignores EXEC, issues per wave).
static __device__ __forceinline__ void tdm_load_1d(unsigned lds_addr, const void* gsrc, unsigned bytes) {
  unsigned long long ga = (unsigned long long)(uintptr_t)gsrc;
  unsigned n = bytes >> 2;                       // 4-byte elements (n <= 65535)
  u32x4 g0;
  g0[0] = 1u;                                    // count=1: valid user descriptor
  g0[1] = lds_addr;                              // LDS byte address
  g0[2] = (unsigned)ga;                          // global_addr[31:0]
  g0[3] = (unsigned)((ga >> 32) & 0x01FFFFFFull) | (2u << 30);   // addr[56:32], type=2
  i32x8 g1;
  g1[0] = (int)(2u << 16);                       // workgroup_mask=0, data_size=2 (4B)
  g1[1] = (int)((n & 0xFFFFu) << 16);            // tensor_dim0[15:0]
  g1[2] = (int)(((n >> 16) & 0xFFFFu) | (1u << 16));  // tensor_dim0[31:16], tensor_dim1=1
  g1[3] = (int)((n & 0xFFFFu) << 16);            // tile_dim0 = n
  g1[4] = 0;                                     // tile_dim1/2 unused
  g1[5] = (int)n;                                // tensor_dim0_stride (irrelevant, 1 row)
  g1[6] = 0; g1[7] = 0;
  i32x4 gz; gz[0] = 0; gz[1] = 0; gz[2] = 0; gz[3] = 0;
#if TDM_6ARG
  i32x8 z8; z8[0]=0; z8[1]=0; z8[2]=0; z8[3]=0; z8[4]=0; z8[5]=0; z8[6]=0; z8[7]=0;
  __builtin_amdgcn_tensor_load_to_lds(g0, g1, gz, gz, z8, 0);
#else
  __builtin_amdgcn_tensor_load_to_lds(g0, g1, gz, gz, 0);
#endif
}

// ---------- f32 -> bf16 bulk convert ----------
__global__ void cvt_bf16_kernel(const float* __restrict__ in, unsigned short* __restrict__ out, int n) {
  int i = blockIdx.x * blockDim.x + threadIdx.x;
  if (i < n) out[i] = f32_bf16(in[i]);
}

// ---------- embedding gather + convert ----------
__global__ void gather_emb_kernel(const int* __restrict__ sent, const float* __restrict__ emb,
                                  unsigned short* __restrict__ xb) {
  int l = blockIdx.x;
  int r = sent[l];
  const float* src = emb + (size_t)r * E_DIM;
  unsigned short* dst = xb + (size_t)l * E_DIM;
  for (int e = threadIdx.x; e < E_DIM; e += blockDim.x) dst[e] = f32_bf16(src[e]);
}

// ---------- counters init (ws is poisoned; must be zeroed every call) ----------
__global__ void zero_counters_kernel(int* __restrict__ c) {
  if (threadIdx.x < 8) c[threadIdx.x] = 0;
}

// ---------- WMMA GEMM: out[M,N] = X[M,K] * W[N,K]^T + bias ----------
// grid = (M/32, N/128), block = 256 (8 waves). TDM DMAs the 32xK X strip into LDS;
// wave w owns n0 = by*128 + w*16, computing two 16x16 tiles sharing one B fragment.
union AFrag { v16bf v; uint4 q[2]; };

__global__ void __launch_bounds__(256) wmma_gemm_xg(
    const unsigned short* __restrict__ X,   // [M,K] bf16 row-major
    const unsigned short* __restrict__ W,   // [N,K] bf16 row-major
    const float* __restrict__ bias,         // [N]
    float* __restrict__ out, int M, int N, int K) {
  __shared__ __align__(16) unsigned short xs[32 * 1024];   // 32 x K strip (K==1024), 64KB

  const int m0 = blockIdx.x * 32;
  if (threadIdx.x < 32) {                      // wave 0 issues the tensor DMA
    tdm_load_1d((unsigned)(uintptr_t)xs, X + (size_t)m0 * K, 32u * 1024u * 2u);
    __builtin_amdgcn_s_wait_tensorcnt(0);
  }
  __syncthreads();

  const int wave = threadIdx.x >> 5;
  const int lane = threadIdx.x & 31;
  const int n0   = blockIdx.y * 128 + wave * 16;
  const int lmod = lane & 15;                // row (A) / col (B) index within tile
  const int hi8  = (lane >> 4) * 8;          // K sub-chunk select per ISA bf16 layout

  const unsigned short* wrow = W + (size_t)(n0 + lmod) * K;   // B column j = n0+lmod
  const float bj = bias[n0 + lmod];

  v8f c0, c1;
#pragma unroll
  for (int r = 0; r < 8; ++r) { c0[r] = bj; c1[r] = bj; }

  for (int k0 = 0; k0 < K; k0 += 32) {
    if ((k0 & 255) == 0) __builtin_prefetch(wrow + k0 + 512, 0, 3);   // global_prefetch_b8
    AFrag a0, a1, b;
    b.q[0]  = *(const uint4*)(&wrow[k0 + hi8]);
    b.q[1]  = *(const uint4*)(&wrow[k0 + 16 + hi8]);
    a0.q[0] = *(const uint4*)(&xs[lmod * 1024 + k0 + hi8]);
    a0.q[1] = *(const uint4*)(&xs[lmod * 1024 + k0 + 16 + hi8]);
    a1.q[0] = *(const uint4*)(&xs[(lmod + 16) * 1024 + k0 + hi8]);
    a1.q[1] = *(const uint4*)(&xs[(lmod + 16) * 1024 + k0 + 16 + hi8]);
    c0 = __builtin_amdgcn_wmma_f32_16x16x32_bf16(false, a0.v, false, b.v, (short)0, c0, false, false);
    c1 = __builtin_amdgcn_wmma_f32_16x16x32_bf16(false, a1.v, false, b.v, (short)0, c1, false, false);
  }

  // D layout: VGPR r, lane l -> M = r + (l>=16)*8, N = l&15
#pragma unroll
  for (int r = 0; r < 8; ++r) {
    out[(size_t)(m0 + r + hi8) * N + n0 + lmod]      = c0[r];
    out[(size_t)(m0 + 16 + r + hi8) * N + n0 + lmod] = c1[r];
  }
}

// ---------- persistent bidirectional LSTM ----------
// 32 workgroups: blocks [0,16) = forward, [16,32) = backward. Each WG owns 32 cells
// (128 gate rows); its 128KB W_hh slice is TDM-DMA'd ONCE into LDS and stays
// resident for all 4096 steps, so per step only the 2KB h vector crosses L2.
// Grid sync = monotonic device-scope atomic counter (+ s_sleep spin); the CDNA5
// cluster barrier is also issued (S_NOP when not dispatched as a cluster).
// h is double-buffered by step parity.
union W8 { uint4 q; __bf16 h[8]; };

__global__ void __launch_bounds__(256) lstm_persistent_kernel(
    const float* __restrict__ xg_f, const float* __restrict__ xg_b,
    const unsigned short* __restrict__ whh_f, const unsigned short* __restrict__ whh_b,
    const float* __restrict__ h0, const float* __restrict__ c0,
    float* __restrict__ hs_f, float* __restrict__ hs_b,
    float* __restrict__ hbuf_f, float* __restrict__ hbuf_b,   // each 2*HDIM floats
    int* __restrict__ counters) {                             // [2]
  extern __shared__ __align__(16) unsigned char smem_raw[];
  unsigned short* wsl = (unsigned short*)smem_raw;            // [128][512] bf16 slice
  float* sh = (float*)(smem_raw + (size_t)ROWS_PER_WG * HDIM * 2);  // [512] h
  float* sg = sh + HDIM;                                      // [128] gate pre-acts
  float* sc = sg + ROWS_PER_WG;                               // [32]  c slice

  const int wg  = blockIdx.x & (NWG - 1);
  const int dir = blockIdx.x >> 4;
  const float* xg = dir ? xg_b : xg_f;
  const unsigned short* whh = dir ? whh_b : whh_f;
  float* hs   = dir ? hs_b : hs_f;
  float* hbuf = dir ? hbuf_b : hbuf_f;
  int* counter = counters + dir;
  const int tid = threadIdx.x;

  // TDM preload: 4 contiguous 32-row gate blocks (32KB each) -> LDS rows
  // lr = gate*32 + cell  <-  global row gate*512 + wg*32 + cell
  if (tid < 32) {
    unsigned ldsb = (unsigned)(uintptr_t)wsl;
    for (int g = 0; g < 4; ++g) {
      const unsigned short* src = whh + ((size_t)g * HDIM + (size_t)wg * CELLS_PER_WG) * HDIM;
      tdm_load_1d(ldsb + (unsigned)g * 32768u, src, 32768u);
    }
    __builtin_amdgcn_s_wait_tensorcnt(0);
  }
  for (int k = tid; k < HDIM; k += 256) sh[k] = h0[dir * HDIM + k];
  if (tid < CELLS_PER_WG) sc[tid] = c0[dir * HDIM + wg * CELLS_PER_WG + tid];
  __syncthreads();

  const int r    = tid >> 1;              // local row 0..127
  const int half = tid & 1;               // dot-product split between lane pairs
  const int kbeg = half * (HDIM / 2);
  const int grow = (r >> 5) * HDIM + wg * CELLS_PER_WG + (r & 31);
  const unsigned short* wrow = wsl + (size_t)r * HDIM;

  for (int s = 0; s < L_SEQ; ++s) {
    const int t = dir ? (L_SEQ - 1 - s) : s;
    float acc = half ? 0.0f : xg[(size_t)t * G4 + grow];
#pragma unroll 4
    for (int k = kbeg; k < kbeg + HDIM / 2; k += 8) {
      W8 w; w.q = *(const uint4*)(wrow + k);
      float4 hA = *(const float4*)(sh + k);
      float4 hB = *(const float4*)(sh + k + 4);
      // mixed-precision FMAs: bf16 weight * f32 h (V_FMA_MIX_F32_BF16 candidates)
      acc = fmaf((float)w.h[0], hA.x, acc); acc = fmaf((float)w.h[1], hA.y, acc);
      acc = fmaf((float)w.h[2], hA.z, acc); acc = fmaf((float)w.h[3], hA.w, acc);
      acc = fmaf((float)w.h[4], hB.x, acc); acc = fmaf((float)w.h[5], hB.y, acc);
      acc = fmaf((float)w.h[6], hB.z, acc); acc = fmaf((float)w.h[7], hB.w, acc);
    }
    acc += __shfl_xor(acc, 1, 32);        // combine the two halves of the dot
    if (!half) sg[r] = acc;
    __syncthreads();

    if (tid < CELLS_PER_WG) {
      float iv = sigmoidf(sg[tid]);
      float fv = sigmoidf(sg[CELLS_PER_WG + tid]);
      float gv = tanhf(sg[2 * CELLS_PER_WG + tid]);
      float ov = sigmoidf(sg[3 * CELLS_PER_WG + tid]);
      float cc = fv * sc[tid] + iv * gv;
      sc[tid] = cc;
      float hh = ov * tanhf(cc);
      int k = wg * CELLS_PER_WG + tid;
      hs[(size_t)t * HDIM + k]          = hh;
      hbuf[((s + 1) & 1) * HDIM + k]    = hh;   // parity double-buffer
    }
    __syncthreads();

    if (tid == 0) {
      __threadfence();                                   // publish hbuf stores
      __hip_atomic_fetch_add(counter, 1, __ATOMIC_RELEASE, __HIP_MEMORY_SCOPE_AGENT);
      __builtin_amdgcn_s_cluster_barrier();              // S_NOP unless cluster dispatch
      while (__hip_atomic_load(counter, __ATOMIC_ACQUIRE, __HIP_MEMORY_SCOPE_AGENT) < NWG * (s + 1))
        __builtin_amdgcn_s_sleep(1);
    }
    __syncthreads();
    for (int k = tid; k < HDIM; k += 256) sh[k] = hbuf[((s + 1) & 1) * HDIM + k];
    __syncthreads();
  }
}

// ---------- output projection: feats[l][tag] = <concat(hf,hb)[l], W_out[tag]> + b ----------
__global__ void __launch_bounds__(192) out_proj_kernel(
    const float* __restrict__ hf, const float* __restrict__ hb,
    const float* __restrict__ W_out, const float* __restrict__ b_out,
    float* __restrict__ feats) {
  const int l = blockIdx.x;
  const int wave = threadIdx.x >> 5, lane = threadIdx.x & 31;   // 6 waves = 6 tags
  const float* wrow = W_out + (size_t)wave * (2 * HDIM);
  float s = 0.0f;
  for (int k = lane; k < HDIM; k += 32) s = fmaf(hf[(size_t)l * HDIM + k], wrow[k], s);
  for (int k = lane; k < HDIM; k += 32) s = fmaf(hb[(size_t)l * HDIM + k], wrow[HDIM + k], s);
  for (int off = 16; off; off >>= 1) s += __shfl_xor(s, off, 32);
  if (lane == 0) feats[l * NTAGS + wave] = s + b_out[wave];
}

// ---------- Viterbi DP + backtrace (single wave; lane n owns tag n) ----------
__global__ void viterbi_kernel(const float* __restrict__ feats, const float* __restrict__ T,
                               int* __restrict__ bps, float* __restrict__ out) {
  __shared__ float Ts[NTAGS * NTAGS];
  const int lane = threadIdx.x;
  for (int i = lane; i < NTAGS * NTAGS; i += 32) Ts[i] = T[i];
  __syncthreads();

  float v = (lane == TAG_START) ? 0.0f : -15000.0f;
  for (int t = 0; t < L_SEQ; ++t) {
    float best = -3.4e38f; int arg = 0;
#pragma unroll
    for (int p = 0; p < NTAGS; ++p) {
      float vp = __shfl(v, p, 32);
      float sc = (lane < NTAGS) ? (Ts[lane * NTAGS + p] + vp) : -3.4e38f;
      if (sc > best) { best = sc; arg = p; }
    }
    float f = (lane < NTAGS) ? feats[t * NTAGS + lane] : 0.0f;
    v = best + f;
    if (lane < NTAGS) bps[t * NTAGS + lane] = arg;
  }
  float last = v + ((lane < NTAGS) ? Ts[TAG_STOP * NTAGS + lane] : 0.0f);
  float bscore = -3.4e38f; int btag = 0;
  for (int n = 0; n < NTAGS; ++n) {
    float ln = __shfl(last, n, 32);
    if (ln > bscore) { bscore = ln; btag = n; }
  }
  if (lane == 0) {
    out[0] = bscore;
    int curr = btag;
    for (int t = L_SEQ - 1; t >= 0; --t) {
      out[1 + t] = (float)curr;
      curr = bps[t * NTAGS + curr];
    }
  }
}

// ---------- host ----------
extern "C" void kernel_launch(void* const* d_in, const int* in_sizes, int n_in,
                              void* d_out, int out_size, void* d_ws, size_t ws_size,
                              hipStream_t stream) {
  (void)in_sizes; (void)n_in; (void)out_size; (void)ws_size;
  const int*   sent  = (const int*)d_in[0];
  const float* emb   = (const float*)d_in[1];
  const float* Wihf  = (const float*)d_in[2];
  const float* Whhf  = (const float*)d_in[3];
  const float* bf_   = (const float*)d_in[4];
  const float* Wihb  = (const float*)d_in[5];
  const float* Whhb  = (const float*)d_in[6];
  const float* bb_   = (const float*)d_in[7];
  const float* h0    = (const float*)d_in[8];
  const float* c0    = (const float*)d_in[9];
  const float* Wout  = (const float*)d_in[10];
  const float* bout  = (const float*)d_in[11];
  const float* T     = (const float*)d_in[12];

  char* p = (char*)d_ws;
  auto alloc = [&](size_t bytes) { void* r = (void*)p; p += (bytes + 255) & ~(size_t)255; return r; };
  unsigned short* xb    = (unsigned short*)alloc((size_t)L_SEQ * E_DIM * 2);
  unsigned short* wihf  = (unsigned short*)alloc((size_t)G4 * E_DIM * 2);
  unsigned short* wihb  = (unsigned short*)alloc((size_t)G4 * E_DIM * 2);
  unsigned short* whhf  = (unsigned short*)alloc((size_t)G4 * HDIM * 2);
  unsigned short* whhb  = (unsigned short*)alloc((size_t)G4 * HDIM * 2);
  float* xgf   = (float*)alloc((size_t)L_SEQ * G4 * 4);
  float* xgb   = (float*)alloc((size_t)L_SEQ * G4 * 4);
  float* hf    = (float*)alloc((size_t)L_SEQ * HDIM * 4);
  float* hb    = (float*)alloc((size_t)L_SEQ * HDIM * 4);
  float* feats = (float*)alloc((size_t)L_SEQ * NTAGS * 4);
  int*   bps   = (int*)alloc((size_t)L_SEQ * NTAGS * 4);
  float* hbf   = (float*)alloc((size_t)2 * HDIM * 4);   // fwd h double-buffer
  float* hbb   = (float*)alloc((size_t)2 * HDIM * 4);   // bwd h double-buffer
  int*   ctrs  = (int*)alloc(64);

  // 0) zero the grid-sync counters (ws is poisoned; must be re-zeroed every call)
  zero_counters_kernel<<<1, 64, 0, stream>>>(ctrs);

  // 1) convert weights to bf16
  {
    int n_ih = G4 * E_DIM, n_hh = G4 * HDIM;
    cvt_bf16_kernel<<<(n_ih + 255) / 256, 256, 0, stream>>>(Wihf, wihf, n_ih);
    cvt_bf16_kernel<<<(n_ih + 255) / 256, 256, 0, stream>>>(Wihb, wihb, n_ih);
    cvt_bf16_kernel<<<(n_hh + 255) / 256, 256, 0, stream>>>(Whhf, whhf, n_hh);
    cvt_bf16_kernel<<<(n_hh + 255) / 256, 256, 0, stream>>>(Whhb, whhb, n_hh);
  }
  // 2) gather embeddings -> bf16
  gather_emb_kernel<<<L_SEQ, 256, 0, stream>>>(sent, emb, xb);

  // 3) input projections via WMMA:  xg = x @ W_ih^T + b   (M=4096, N=2048, K=1024)
  dim3 ggrid(L_SEQ / 32, G4 / 128);
  wmma_gemm_xg<<<ggrid, 256, 0, stream>>>(xb, wihf, bf_, xgf, L_SEQ, G4, E_DIM);
  wmma_gemm_xg<<<ggrid, 256, 0, stream>>>(xb, wihb, bb_, xgb, L_SEQ, G4, E_DIM);

  // 4) persistent recurrences: 16 WGs fwd + 16 WGs bwd, LDS-resident W_hh slices
  lstm_persistent_kernel<<<2 * NWG, 256, LSTM_SMEM, stream>>>(
      xgf, xgb, whhf, whhb, h0, c0, hf, hb, hbf, hbb, ctrs);

  // 5) tag features
  out_proj_kernel<<<L_SEQ, 192, 0, stream>>>(hf, hb, Wout, bout, feats);

  // 6) Viterbi + backtrace -> d_out[0]=score, d_out[1..L]=path
  viterbi_kernel<<<1, 32, 0, stream>>>(feats, T, bps, (float*)d_out);
}
